// GPN_GCN_59442347377123
// MI455X (gfx1250) — matrix operand order repaired and checked by reading, not compiled, not run on hardware
//
#include <hip/hip_runtime.h>
#include <hip/hip_bf16.h>
#include <stdint.h>

typedef __attribute__((ext_vector_type(16))) __bf16 v16bf;
typedef __attribute__((ext_vector_type(8)))  float  v8f;

#define NN_NODES 100000
#define NN_EDGES 1600000
#define F_IN 256
#define HID 128
#define N_CLS 64
#define K_APPNP 10

static __device__ __forceinline__ void atomAddF(float* p, float v) {
    __hip_atomic_fetch_add(p, v, __ATOMIC_RELAXED, __HIP_MEMORY_SCOPE_AGENT);
}

// ---------------- elementwise / graph-prep kernels ----------------

__global__ void k_zero(float* __restrict__ p, size_t n) {
    size_t i = (size_t)blockIdx.x * blockDim.x + threadIdx.x;
    if (i < n) p[i] = 0.0f;
}

__global__ void k_edges32(const long long* __restrict__ row64,
                          const long long* __restrict__ col64,
                          int* __restrict__ rows, int* __restrict__ cols,
                          int nE, int nTot) {
    int e = blockIdx.x * blockDim.x + threadIdx.x;
    if (e >= nTot) return;
    if (e < nE) { rows[e] = (int)row64[e]; cols[e] = (int)col64[e]; }
    else        { int v = e - nE; rows[e] = v; cols[e] = v; }
}

__global__ void k_deg(const int* __restrict__ cols, float* __restrict__ deg, int nTot) {
    int e = blockIdx.x * blockDim.x + threadIdx.x;
    if (e >= nTot) return;
    atomAddF(&deg[cols[e]], 1.0f);
}

__global__ void k_rsqrt(float* __restrict__ d, int n) {
    int i = blockIdx.x * blockDim.x + threadIdx.x;
    if (i >= n) return;
    float v = d[i];
    d[i] = (v > 0.0f) ? rsqrtf(v) : 0.0f;
}

__global__ void k_norm(const int* __restrict__ rows, const int* __restrict__ cols,
                       const float* __restrict__ dinv, float* __restrict__ nrm, int nTot) {
    int e = blockIdx.x * blockDim.x + threadIdx.x;
    if (e >= nTot) return;
    nrm[e] = dinv[rows[e]] * dinv[cols[e]];
}

// W[K x Nn] fp32 row-major -> Wt[Nn x K] bf16 (transposed, contiguous-K per column)
__global__ void k_cvt_wT(const float* __restrict__ W, __bf16* __restrict__ Wt, int K, int Nn) {
    int i = blockIdx.x * blockDim.x + threadIdx.x;
    if (i >= K * Nn) return;
    int kk = i / Nn, n = i % Nn;
    Wt[(size_t)n * K + kk] = (__bf16)W[i];
}

// ---------------- normalized scatter-add propagation ----------------
// hout[col[e], :] += norm[e] * hin[row[e], :]   (F = 4 << shift features)
__global__ void k_prop(const float* __restrict__ hin, float* __restrict__ hout,
                       const int* __restrict__ rows, const int* __restrict__ cols,
                       const float* __restrict__ nrm, unsigned nTot, int shift) {
    unsigned idx = blockIdx.x * blockDim.x + threadIdx.x;
    unsigned total = nTot << shift;
    if (idx >= total) return;
    unsigned e = idx >> shift;
    unsigned c = (idx & ((1u << shift) - 1u)) * 4u;
    int F = 4 << shift;
    int r = rows[e], cl = cols[e];
    float w = nrm[e];
    const float4 v = *(const float4*)(hin + (size_t)r * F + c);
    float* dst = hout + (size_t)cl * F + c;
    atomAddF(dst + 0, v.x * w);
    atomAddF(dst + 1, v.y * w);
    atomAddF(dst + 2, v.z * w);
    atomAddF(dst + 3, v.w * w);
}

__global__ void k_bias_act(float* __restrict__ h, const float* __restrict__ b,
                           unsigned n, int mask, int relu) {
    unsigned i = blockIdx.x * blockDim.x + threadIdx.x;
    if (i >= n) return;
    float v = h[i] + b[i & mask];
    h[i] = relu ? fmaxf(v, 0.0f) : v;
}

__global__ void k_copy(float* __restrict__ dst, const float* __restrict__ src, unsigned n) {
    unsigned i = blockIdx.x * blockDim.x + threadIdx.x;
    if (i < n) dst[i] = src[i];
}

// z = 0.9*tmp + 0.1*ev
__global__ void k_axpy(float* __restrict__ z, const float* __restrict__ t,
                       const float* __restrict__ ev, unsigned n) {
    unsigned i = blockIdx.x * blockDim.x + threadIdx.x;
    if (i < n) z[i] = 0.9f * t[i] + 0.1f * ev[i];
}

// ---------------- WMMA GEMM: C[M x (NT*16)] = A[M x K](fp32->bf16) @ Wt' ----------------
// Wt is bf16, stored transposed [N x K]. One wave -> 16 output rows x NT 16-col tiles.
template <int K, int NT>
__global__ __launch_bounds__(256) void gemm_wmma(const float* __restrict__ A,
                                                 const __bf16* __restrict__ Wt,
                                                 float* __restrict__ C,
                                                 const float* __restrict__ bias,
                                                 int M, int relu) {
    const int NNc = NT * 16;
    int wave = threadIdx.x >> 5;
    int lane = threadIdx.x & 31;
    int tile = blockIdx.x * (blockDim.x >> 5) + wave;
    if (tile * 16 >= M) return;              // wave-uniform: EXEC stays all-ones
    int half = lane >> 4;                    // K half selector
    int mn   = lane & 15;                    // row (A) / col (B,C) within tile

    const float* arow = A + (size_t)(tile * 16 + mn) * K;
    v8f acc[NT] = {};

    for (int k = 0; k < K; k += 32) {
        // A fragment: 16x32 bf16. Lane (half,m): K = 16h..16h+15 contiguous (matches B)
        const float* ap = arow + k + 16 * half;
        float4 p0 = *(const float4*)(ap + 0);
        float4 p1 = *(const float4*)(ap + 4);
        float4 p2 = *(const float4*)(ap + 8);
        float4 p3 = *(const float4*)(ap + 12);
        v16bf a;
        a[0]  = (__bf16)p0.x; a[1]  = (__bf16)p0.y; a[2]  = (__bf16)p0.z; a[3]  = (__bf16)p0.w;
        a[4]  = (__bf16)p1.x; a[5]  = (__bf16)p1.y; a[6]  = (__bf16)p1.z; a[7]  = (__bf16)p1.w;
        a[8]  = (__bf16)p2.x; a[9]  = (__bf16)p2.y; a[10] = (__bf16)p2.z; a[11] = (__bf16)p2.w;
        a[12] = (__bf16)p3.x; a[13] = (__bf16)p3.y; a[14] = (__bf16)p3.z; a[15] = (__bf16)p3.w;
#pragma unroll
        for (int nt = 0; nt < NT; ++nt) {
            // B fragment: 32x16, lane half holds K = 16h..16h+15 contiguous -> one 32B load
            v16bf b = *(const v16bf*)(Wt + (size_t)(nt * 16 + mn) * K + k + 16 * half);
            acc[nt] = __builtin_amdgcn_wmma_f32_16x16x32_bf16(
                false, a, false, b, (short)0, acc[nt], false, false);
        }
    }

#pragma unroll
    for (int nt = 0; nt < NT; ++nt) {
        int n = nt * 16 + mn;
        float bv = bias ? bias[n] : 0.0f;
#pragma unroll
        for (int r = 0; r < 8; ++r) {        // C layout: vgpr r -> row r + 8*half
            float v = acc[nt][r] + bv;
            if (relu) v = fmaxf(v, 0.0f);
            C[(size_t)(tile * 16 + 8 * half + r) * NNc + n] = v;
        }
    }
}

// ---------------- row-wise log_softmax over 64 classes (wave per row) ----------------
__global__ void k_logsoftmax(const float* __restrict__ z, float* __restrict__ out, int nRows) {
    int wave = threadIdx.x >> 5, lane = threadIdx.x & 31;
    int rowi = blockIdx.x * (blockDim.x >> 5) + wave;
    if (rowi >= nRows) return;
    const float2 v = *(const float2*)(z + (size_t)rowi * 64 + lane * 2);
    float m = fmaxf(v.x, v.y);
    for (int o = 16; o > 0; o >>= 1) m = fmaxf(m, __shfl_xor(m, o, 32));
    float s = expf(v.x - m) + expf(v.y - m);
    for (int o = 16; o > 0; o >>= 1) s += __shfl_xor(s, o, 32);
    float ls = m + logf(s);
    float2 r; r.x = v.x - ls; r.y = v.y - ls;
    *(float2*)(out + (size_t)rowi * 64 + lane * 2) = r;
}

// ---------------- driver ----------------
extern "C" void kernel_launch(void* const* d_in, const int* in_sizes, int n_in,
                              void* d_out, int out_size, void* d_ws, size_t ws_size,
                              hipStream_t stream) {
    (void)in_sizes; (void)n_in; (void)out_size; (void)ws_size;
    const float*     x  = (const float*)d_in[0];
    const long long* ei = (const long long*)d_in[1];
    const float* W1 = (const float*)d_in[2];
    const float* b1 = (const float*)d_in[3];
    const float* W2 = (const float*)d_in[4];
    const float* b2 = (const float*)d_in[5];
    const float* Wc = (const float*)d_in[6];
    const float* bc = (const float*)d_in[7];
    const float* We = (const float*)d_in[8];
    const float* be = (const float*)d_in[9];
    float* out = (float*)d_out;

    const int N = NN_NODES, nE = NN_EDGES, nTot = nE + N;

    char* ws = (char*)d_ws;
    size_t off = 0;
    auto carve = [&](size_t bytes) -> char* {
        char* p = ws + off;
        off += (bytes + 255) & ~(size_t)255;
        return p;
    };
    float*  dinv = (float*)carve((size_t)N * 4);       // deg -> dinv in place
    int*    rows = (int*)  carve((size_t)nTot * 4);
    int*    cols = (int*)  carve((size_t)nTot * 4);
    float*  nrm  = (float*)carve((size_t)nTot * 4);
    __bf16* wt1  = (__bf16*)carve((size_t)F_IN * HID * 2);
    __bf16* wt2  = (__bf16*)carve((size_t)HID * HID * 2);
    __bf16* wtc  = (__bf16*)carve((size_t)HID * N_CLS * 2);
    __bf16* wte  = (__bf16*)carve((size_t)N_CLS * N_CLS * 2);
    float*  bufA = (float*)carve((size_t)N * HID * 4);
    float*  bufB = (float*)carve((size_t)N * HID * 4);
    float* logits = bufA;
    float* ev     = bufA + (size_t)N * N_CLS;
    float* z      = bufB;
    float* tmp    = bufB + (size_t)N * N_CLS;

    dim3 blk(256);
    auto g1 = [](size_t n) { return dim3((unsigned)((n + 255) / 256)); };

    // Graph prep: int32 edges (+self loops), deg on targets, dinv, edge norms
    k_edges32<<<g1(nTot), blk, 0, stream>>>(ei, ei + nE, rows, cols, nE, nTot);
    k_zero   <<<g1(N),    blk, 0, stream>>>(dinv, (size_t)N);
    k_deg    <<<g1(nTot), blk, 0, stream>>>(cols, dinv, nTot);
    k_rsqrt  <<<g1(N),    blk, 0, stream>>>(dinv, N);
    k_norm   <<<g1(nTot), blk, 0, stream>>>(rows, cols, dinv, nrm, nTot);

    // Weights -> bf16 transposed
    k_cvt_wT<<<g1(F_IN * HID),   blk, 0, stream>>>(W1, wt1, F_IN, HID);
    k_cvt_wT<<<g1(HID * HID),    blk, 0, stream>>>(W2, wt2, HID, HID);
    k_cvt_wT<<<g1(HID * N_CLS),  blk, 0, stream>>>(Wc, wtc, HID, N_CLS);
    k_cvt_wT<<<g1(N_CLS * N_CLS),blk, 0, stream>>>(We, wte, N_CLS, N_CLS);

    const int tilesM = N / 16;                    // 6250, exact
    dim3 gG((tilesM + 7) / 8);                    // 8 waves per block

    // GCN layer 1: (x@W1) -> prop -> +b1, relu
    gemm_wmma<256, 8><<<gG, blk, 0, stream>>>(x, wt1, bufA, nullptr, N, 0);
    k_zero<<<g1((size_t)N * HID), blk, 0, stream>>>(bufB, (size_t)N * HID);
    k_prop<<<g1((size_t)nTot * 32), blk, 0, stream>>>(bufA, bufB, rows, cols, nrm, nTot, 5);
    k_bias_act<<<g1((size_t)N * HID), blk, 0, stream>>>(bufB, b1, (unsigned)(N * HID), HID - 1, 1);

    // GCN layer 2
    gemm_wmma<128, 8><<<gG, blk, 0, stream>>>(bufB, wt2, bufA, nullptr, N, 0);
    k_zero<<<g1((size_t)N * HID), blk, 0, stream>>>(bufB, (size_t)N * HID);
    k_prop<<<g1((size_t)nTot * 32), blk, 0, stream>>>(bufA, bufB, rows, cols, nrm, nTot, 5);
    k_bias_act<<<g1((size_t)N * HID), blk, 0, stream>>>(bufB, b2, (unsigned)(N * HID), HID - 1, 1);

    // logits = h@Wc + bc ; ev = relu(logits@We + be)
    gemm_wmma<128, 4><<<gG, blk, 0, stream>>>(bufB, wtc, logits, bc, N, 0);
    gemm_wmma<64, 4><<<gG, blk, 0, stream>>>(logits, wte, ev, be, N, 1);

    // APPNP: z = ev; 10x { z = 0.9*prop(z) + 0.1*ev }
    k_copy<<<g1((size_t)N * N_CLS), blk, 0, stream>>>(z, ev, (unsigned)(N * N_CLS));
    for (int it = 0; it < K_APPNP; ++it) {
        k_zero<<<g1((size_t)N * N_CLS), blk, 0, stream>>>(tmp, (size_t)N * N_CLS);
        k_prop<<<g1((size_t)nTot * 16), blk, 0, stream>>>(z, tmp, rows, cols, nrm, nTot, 4);
        k_axpy<<<g1((size_t)N * N_CLS), blk, 0, stream>>>(z, tmp, ev, (unsigned)(N * N_CLS));
    }

    // log_softmax rows -> d_out
    k_logsoftmax<<<dim3((N + 7) / 8), blk, 0, stream>>>(z, out, N);
}
